// YOLOSCEMALoss_14628658610932
// MI455X (gfx1250) — compile-verified, matrix operand order checked by reading, not run on hardware
//
#include <hip/hip_runtime.h>
#include <hip/hip_bf16.h>

typedef _Float16 v16h __attribute__((ext_vector_type(16)));
typedef float    v8f  __attribute__((ext_vector_type(8)));

#define NUM_CLASSES 80
#define BATCH 16
#define NUM_T 256
#define EPSF 1e-7f
#define T0 25   /* 6400/256 */
#define T1 7    /* ceil(1600/256) */
#define T2 2    /* ceil(400/256)  */
#define TILES (T0 + T1 + T2)

__device__ __forceinline__ float softplusf(float z) {
    // logaddexp(0, z), numerically stable
    return fmaxf(z, 0.0f) + log1pf(expf(-fabsf(z)));
}
__device__ __forceinline__ float sigmoidf_(float z) {
    return 1.0f / (1.0f + expf(-z));
}

// ---------------------------------------------------------------------------
// Kernel 0: bucket the 256 targets by batch index (deterministic, in order).
// ---------------------------------------------------------------------------
__global__ void yolo_prep(const float* __restrict__ tboxes,
                          const int* __restrict__ tbatch,
                          const int* __restrict__ tcls,
                          int* __restrict__ cnt,
                          float4* __restrict__ wtb,
                          int* __restrict__ wtc) {
    int b = threadIdx.x;
    if (b >= BATCH) return;
    int n = 0;
    for (int t = 0; t < NUM_T; ++t) {
        if (tbatch[t] == b) {
            const float* p = tboxes + 4 * t;
            wtb[b * NUM_T + n] = make_float4(p[0], p[1], p[2], p[3]);
            wtc[b * NUM_T + n] = tcls[t];
            ++n;
        }
    }
    cnt[b] = n;
}

// ---------------------------------------------------------------------------
// Kernel 1: per-anchor losses. grid = (34 tiles, 16 batches), 256 threads.
// Phase 1: decode + IoU matching + CIoU (thread = anchor).
// Phase 2: class pass; softplus row-sums via v_wmma_f32_16x16x32_f16
//          (wave = 16 anchors per group, 2 groups of the wave's 32 anchors).
// ---------------------------------------------------------------------------
__global__ __launch_bounds__(256)
void yolo_main(const float* __restrict__ cls0, const float* __restrict__ reg0,
               const float* __restrict__ cls1, const float* __restrict__ reg1,
               const float* __restrict__ cls2, const float* __restrict__ reg2,
               const int* __restrict__ cnt, const float4* __restrict__ wtb,
               const int* __restrict__ wtc, float* __restrict__ partials) {
    __shared__ float4 s_tbox[NUM_T];
    __shared__ int    s_tcls[NUM_T];
    __shared__ int    s_cnt;
    __shared__ float  s_sums[8 * 32];   // [wave][32 anchors] softplus row-sums
    __shared__ float  s_wred[8][4];

    const int tid = threadIdx.x;
    const int b   = blockIdx.y;
    const int sx  = blockIdx.x;

    int A, W, bxs;
    float stride;
    const float *cls, *reg;
    if (sx < T0)           { bxs = sx;           A = 6400; W = 80; stride = 8.0f;  cls = cls0; reg = reg0; }
    else if (sx < T0 + T1) { bxs = sx - T0;      A = 1600; W = 40; stride = 16.0f; cls = cls1; reg = reg1; }
    else                   { bxs = sx - T0 - T1; A = 400;  W = 20; stride = 32.0f; cls = cls2; reg = reg2; }

    if (tid == 0) s_cnt = cnt[b];
    __syncthreads();
    const int tc = s_cnt;
    if (tid < tc) {
        s_tbox[tid] = wtb[b * NUM_T + tid];
        s_tcls[tid] = wtc[b * NUM_T + tid];
    }
    __syncthreads();

    // ---------------- Phase 1: decode + match ----------------
    const int  a     = bxs * 256 + tid;
    const bool valid = (a < A);
    const int  ac    = valid ? a : 0;

    const float* rp = reg + (size_t)b * 4 * A;
    float r0 = rp[ac], r1 = rp[A + ac], r2 = rp[2 * A + ac], r3 = rp[3 * A + ac];
    float gx = (float)(ac % W), gy = (float)(ac / W);
    float pcx = (sigmoidf_(r0) + gx) * stride;
    float pcy = (sigmoidf_(r1) + gy) * stride;
    float pw0 = expf(r2) * stride;
    float ph0 = expf(r3) * stride;
    float px1 = pcx - 0.5f * pw0, py1 = pcy - 0.5f * ph0;
    float px2 = pcx + 0.5f * pw0, py2 = pcy + 0.5f * ph0;
    float pw = px2 - px1, ph = py2 - py1;
    float areaA = pw * ph;

    float best_iou = -1.0f;
    int   best = 0;
    for (int t = 0; t < tc; ++t) {
        float4 g = s_tbox[t];
        float iw = fmaxf(fminf(px2, g.z) - fmaxf(px1, g.x), 0.0f);
        float ih = fmaxf(fminf(py2, g.w) - fmaxf(py1, g.y), 0.0f);
        float inter = iw * ih;
        float areaB = (g.z - g.x) * (g.w - g.y);
        float iou = inter / (areaA + areaB - inter + EPSF);
        if (iou > best_iou) { best_iou = iou; best = t; }
    }
    const bool pos = valid && (best_iou > 0.5f);
    const int  t_a = (tc > 0) ? s_tcls[best] : 0;

    float boxc = 0.0f, nposc = 0.0f;
    if (pos) {
        float4 g = s_tbox[best];
        float gw = g.z - g.x, gh = g.w - g.y;
        float iw = fmaxf(fminf(px2, g.z) - fmaxf(px1, g.x), 0.0f);
        float ih = fmaxf(fminf(py2, g.w) - fmaxf(py1, g.y), 0.0f);
        float inter = iw * ih;
        float uni   = pw * ph + gw * gh - inter + EPSF;
        float iou   = inter / uni;
        float ccw = fmaxf(px2, g.z) - fminf(px1, g.x);
        float cch = fmaxf(py2, g.w) - fminf(py1, g.y);
        float c2  = ccw * ccw + cch * cch + EPSF;
        float dx  = (px1 + px2) * 0.5f - (g.x + g.z) * 0.5f;
        float dy  = (py1 + py2) * 0.5f - (g.y + g.w) * 0.5f;
        float rho2 = dx * dx + dy * dy;
        const float kpi = 4.0f / (3.14159265358979323846f * 3.14159265358979323846f);
        float da = atanf(gw / (gh + EPSF)) - atanf(pw / (ph + EPSF));
        float v  = kpi * da * da;
        float alpha = v / (v - iou + 1.0f + EPSF);
        float ciou  = iou - rho2 / c2 - alpha * v;
        boxc = 1.0f - ciou;
        nposc = 1.0f;
    }

    // ---------------- Phase 2: class pass with WMMA ----------------
    const int lane = tid & 31;
    const int wid  = tid >> 5;
    const int wave_base = bxs * 256 + wid * 32;

    float posf = pos ? 1.0f : 0.0f;
    float valf = valid ? 1.0f : 0.0f;
    float posf_o = __shfl_xor(posf, 16, 32);
    float valf_o = __shfl_xor(valf, 16, 32);
    int   t_o    = __shfl_xor(t_a, 16, 32);

    // f16 16x32 A-matrix K mapping: lanes 0-15 hold K in {0..7,16..23},
    // lanes 16-31 hold K in {8..15,24..31}  =>  kk = kbase + e + (e>=8 ? 8 : 0)
    const int kbase = (lane >> 4) * 8;
    const float* cp0 = cls + (size_t)b * NUM_CLASSES * A;

    v16h ones;
    #pragma unroll
    for (int e = 0; e < 16; ++e) ones[e] = (_Float16)1.0f;

    float cls_acc = 0.0f, obj_acc = 0.0f;

    #pragma unroll
    for (int g = 0; g < 2; ++g) {
        const bool  own   = ((lane >> 4) == g);
        const float pos_g = own ? posf : posf_o;
        const float val_g = own ? valf : valf_o;
        const int   t_g   = own ? t_a : t_o;

        int a_g = wave_base + g * 16 + (lane & 15);
        int agc = (a_g < A) ? a_g : 0;           // clamp keeps loads in bounds
        const float* cp = cp0 + agc;

        float maxz = -1e30f, zt = -1e30f;
        v8f acc = {0.f, 0.f, 0.f, 0.f, 0.f, 0.f, 0.f, 0.f};

        #pragma unroll
        for (int ch = 0; ch < 3; ++ch) {          // 3 x K=32 covers 80 classes
            v16h av;
            #pragma unroll
            for (int e = 0; e < 16; ++e) {
                if (ch == 2 && e >= 8) {          // classes 80..95: pad (sp=0)
                    av[e] = (_Float16)0.0f;
                    continue;
                }
                int kk = kbase + e + ((e >= 8) ? 8 : 0);
                int cc = ch * 32 + kk;            // global class, < 80 here
                float z = cp[(size_t)cc * A];
                maxz = fmaxf(maxz, z);
                zt   = (cc == t_g) ? z : zt;
                av[e] = (_Float16)softplusf(z);
            }
            // D[m,n] += sum_k A[m,k] * 1  -> every column of D = row sums
            acc = __builtin_amdgcn_wmma_f32_16x16x32_f16(
                false, av, false, ones, (short)0, acc, false, false);
        }

        // combine the lane pair (each covered half the classes)
        maxz = fmaxf(maxz, __shfl_xor(maxz, 16, 32));
        zt   = fmaxf(zt,   __shfl_xor(zt,   16, 32));

        // D column 0: lane 0 holds rows 0-7 in c[0..7], lane 16 rows 8-15
        if (lane == 0) {
            #pragma unroll
            for (int r = 0; r < 8; ++r) s_sums[wid * 32 + g * 16 + r] = acc[r];
        } else if (lane == 16) {
            #pragma unroll
            for (int r = 0; r < 8; ++r) s_sums[wid * 32 + g * 16 + 8 + r] = acc[r];
        }
        __syncthreads();
        if (lane < 16) {
            float sum_sp = s_sums[wid * 32 + g * 16 + lane];
            // obj: bce(max logit, pos);  cls: sum_c softplus(z) - z_target
            obj_acc += val_g * (softplusf(maxz) - maxz * pos_g);
            cls_acc += val_g * pos_g * (sum_sp - zt);
        }
        __syncthreads();
    }

    // ---------------- block reduction (deterministic) ----------------
    float v0 = boxc, v1 = nposc, v2 = cls_acc, v3 = obj_acc;
    #pragma unroll
    for (int off = 16; off >= 1; off >>= 1) {
        v0 += __shfl_xor(v0, off, 32);
        v1 += __shfl_xor(v1, off, 32);
        v2 += __shfl_xor(v2, off, 32);
        v3 += __shfl_xor(v3, off, 32);
    }
    if (lane == 0) {
        s_wred[wid][0] = v0; s_wred[wid][1] = v1;
        s_wred[wid][2] = v2; s_wred[wid][3] = v3;
    }
    __syncthreads();
    if (tid < 4) {
        float s = 0.0f;
        #pragma unroll
        for (int w = 0; w < 8; ++w) s += s_wred[w][tid];
        partials[((size_t)blockIdx.y * TILES + blockIdx.x) * 4 + tid] = s;
    }
}

// ---------------------------------------------------------------------------
// Kernel 2: fixed-order final reduction + gains.
// ---------------------------------------------------------------------------
__global__ void yolo_finalize(const float* __restrict__ partials,
                              float* __restrict__ out) {
    __shared__ float s12[12];
    const int tid = threadIdx.x;
    const int lo[3] = {0, T0, T0 + T1};
    const int hi[3] = {T0, T0 + T1, TILES};
    if (tid < 12) {
        int s = tid >> 2, q = tid & 3;
        float acc = 0.0f;
        for (int by = 0; by < BATCH; ++by)
            for (int bx = lo[s]; bx < hi[s]; ++bx)
                acc += partials[((size_t)by * TILES + bx) * 4 + q];
        s12[tid] = acc;
    }
    __syncthreads();
    if (tid == 0) {
        const float Atab[3] = {6400.0f, 1600.0f, 400.0f};
        float total = 0.0f;
        for (int s = 0; s < 3; ++s) {
            float box_s = s12[s * 4 + 0];
            float npos  = s12[s * 4 + 1];
            float cls_s = s12[s * 4 + 2];
            float obj_s = s12[s * 4 + 3];
            float nposm = fmaxf(npos, 1.0f);
            float box_loss = box_s / nposm;
            float cls_loss = cls_s / (nposm * (float)NUM_CLASSES);
            float obj_loss = obj_s / ((float)BATCH * Atab[s]);
            total += 7.5f * box_loss + 0.5f * cls_loss + 1.0f * obj_loss;
        }
        out[0] = total;
    }
}

// ---------------------------------------------------------------------------
extern "C" void kernel_launch(void* const* d_in, const int* in_sizes, int n_in,
                              void* d_out, int out_size, void* d_ws, size_t ws_size,
                              hipStream_t stream) {
    (void)in_sizes; (void)n_in; (void)out_size; (void)ws_size;
    // setup_inputs() order: cls_p0, reg_p0, cls_p1, reg_p1, cls_p2, reg_p2,
    //                       tboxes, t_batch, t_cls
    const float* cls0 = (const float*)d_in[0];
    const float* reg0 = (const float*)d_in[1];
    const float* cls1 = (const float*)d_in[2];
    const float* reg1 = (const float*)d_in[3];
    const float* cls2 = (const float*)d_in[4];
    const float* reg2 = (const float*)d_in[5];
    const float* tboxes = (const float*)d_in[6];
    const int*   tbatch = (const int*)d_in[7];
    const int*   tcls   = (const int*)d_in[8];

    // workspace layout (floats): [0,2176) partials | [2176,2192) cnt(int) |
    // [2192,18576) bucketed boxes (float4) | [18576,22672) bucketed cls (int)
    float*  wsf      = (float*)d_ws;
    float*  partials = wsf;
    int*    cnt      = (int*)(wsf + 2176);
    float4* wtb      = (float4*)(wsf + 2192);          // 16B-aligned offset
    int*    wtc      = (int*)(wsf + 2192 + 16384);

    yolo_prep<<<1, 16, 0, stream>>>(tboxes, tbatch, tcls, cnt, wtb, wtc);
    yolo_main<<<dim3(TILES, BATCH), 256, 0, stream>>>(
        cls0, reg0, cls1, reg1, cls2, reg2, cnt, wtb, wtc, partials);
    yolo_finalize<<<1, 32, 0, stream>>>(partials, (float*)d_out);
}
